// GIN_54743653155378
// MI455X (gfx1250) — matrix operand order repaired and checked by reading, not compiled, not run on hardware
//
#include <hip/hip_runtime.h>

// ---------------------------------------------------------------------------
// GIN forward on gfx1250 (MI455X): 9 bf16 WMMA GEMMs + edge scatter-add.
// WMMA: v_wmma_f32_16x16x32_bf16, f32 accumulate. Weights pre-packed into
// B-fragment order so B loads are coalesced global_load_b128s.
// v2: double-buffered LDS A staging (1 barrier / K-step), B fragments
// register-prefetched so the 4 WMMAs issue back-to-back, strength-reduced
// B pointers.
// ---------------------------------------------------------------------------

typedef __attribute__((ext_vector_type(16))) __bf16 v16bf;
typedef __attribute__((ext_vector_type(8)))  __bf16 v8bf;
typedef __attribute__((ext_vector_type(4)))  __bf16 v4bf;
typedef __attribute__((ext_vector_type(8)))  float  v8f;

// --------------------------------------------------------------------------
// Pack f32 weight [K][N] into bf16 WMMA B-fragment order:
// blocks are (nb major, kb minor); within a block: lane 0..31, elem 0..15,
//   value = W[kb*32 + (lane/16)*16 + e][nb*16 + (lane%16)]
// so in the GEMM a lane's whole 16-half fragment is 32 contiguous bytes.
// --------------------------------------------------------------------------
__global__ void pack_weight_bf16(const float* __restrict__ W,
                                 __bf16* __restrict__ out, int K, int N) {
  int t = blockIdx.x * blockDim.x + threadIdx.x;
  int total = K * N;
  if (t >= total) return;
  int kbc = K >> 5;                 // K / 32
  int blk = t >> 9;                 // 512 elements per 32x16 block
  int within = t & 511;
  int lane = within >> 4;
  int e = within & 15;
  int nb = blk / kbc;
  int kb = blk - nb * kbc;
  int k = (kb << 5) + ((lane >> 4) << 4) + e;
  int n = (nb << 4) + (lane & 15);
  out[t] = (__bf16)W[k * N + n];
}

// --------------------------------------------------------------------------
// C[M,Ncols] = act(A[M,K] @ W[K,Ncols] + bias) (+ optional residual)
// block = 128 threads = 4 waves; block tile 64x64; wave tile 16x64.
// --------------------------------------------------------------------------
__global__ __launch_bounds__(128) void gemm_wmma_bf16(
    const float* __restrict__ A, const __bf16* __restrict__ Wp,
    const float* __restrict__ bias, const float* __restrict__ resid,
    float* __restrict__ C, int M, int K, int Ncols, int relu) {
  // double-buffered: 64 rows x 32 k-halves, 80B row stride (conflict-free
  // b128 fragment reads: 16 row-starts land on 16 disjoint 4-bank groups)
  __shared__ __bf16 As[2][64][40];

  const int tid  = threadIdx.x;
  const int lane = tid & 31;
  const int wv   = tid >> 5;
  const int half = lane >> 4;
  const int ln   = lane & 15;
  const int row0 = blockIdx.x * 64;
  const int col0 = blockIdx.y * 64;
  const int kbc  = K >> 5;
  const int nb0  = col0 >> 4;
  const v16bf* __restrict__ Wv = (const v16bf*)Wp;

  // per-wave persistent B-fragment pointers; advance by 32 v16bf per K-step
  const v16bf* wf0 = Wv + (size_t)((nb0 + 0) * kbc) * 32 + lane;
  const v16bf* wf1 = Wv + (size_t)((nb0 + 1) * kbc) * 32 + lane;
  const v16bf* wf2 = Wv + (size_t)((nb0 + 2) * kbc) * 32 + lane;
  const v16bf* wf3 = Wv + (size_t)((nb0 + 3) * kbc) * 32 + lane;

  v8f acc[4] = {};

  // stage A tile kb into LDS buffer buf (64x32 f32 -> bf16)
  auto stage = [&](int kb, int buf) {
    const int k0 = kb << 5;
#pragma unroll
    for (int j = 0; j < 4; ++j) {
      int idx4 = j * 128 + tid;          // 0..511 float4 slots
      int r  = idx4 >> 3;                // row in tile
      int c4 = idx4 & 7;                 // float4 within the 32-wide row
      int gm = row0 + r;
      float4 v = make_float4(0.f, 0.f, 0.f, 0.f);
      if (gm < M)
        v = *(const float4*)(A + (size_t)gm * K + k0 + c4 * 4);
      v4bf o;
      o[0] = (__bf16)v.x; o[1] = (__bf16)v.y;
      o[2] = (__bf16)v.z; o[3] = (__bf16)v.w;
      *(v4bf*)&As[buf][r][c4 * 4] = o;
    }
  };

  stage(0, 0);
  __syncthreads();

  int buf = 0;
  for (int kb = 0; kb < kbc; ++kb) {
    // ---- prefetch all 4 B fragments (8 coalesced global_load_b128s) ----
    v16bf b0 = *wf0;
    v16bf b1 = *wf1;
    v16bf b2 = *wf2;
    v16bf b3 = *wf3;
    wf0 += 32; wf1 += 32; wf2 += 32; wf3 += 32;

    // ---- A fragment from current LDS buffer ----
    // lanes 0-15: K=0-15, lanes 16-31: K=16-31
    const __bf16* arow = &As[buf][wv * 16 + ln][0];
    v8bf lo = *(const v8bf*)(arow + half * 16);
    v8bf hi = *(const v8bf*)(arow + half * 16 + 8);
    v16bf afrag;
#pragma unroll
    for (int i = 0; i < 8; ++i) { afrag[i] = lo[i]; afrag[i + 8] = hi[i]; }

    // ---- stage next K tile into the other buffer (overlaps WMMAs) ----
    if (kb + 1 < kbc) stage(kb + 1, buf ^ 1);

    // ---- 4 back-to-back WMMAs against register-resident fragments ----
    acc[0] = __builtin_amdgcn_wmma_f32_16x16x32_bf16(
        false, afrag, false, b0, (short)0, acc[0], false, false);
    acc[1] = __builtin_amdgcn_wmma_f32_16x16x32_bf16(
        false, afrag, false, b1, (short)0, acc[1], false, false);
    acc[2] = __builtin_amdgcn_wmma_f32_16x16x32_bf16(
        false, afrag, false, b2, (short)0, acc[2], false, false);
    acc[3] = __builtin_amdgcn_wmma_f32_16x16x32_bf16(
        false, afrag, false, b3, (short)0, acc[3], false, false);

    __syncthreads();
    buf ^= 1;
  }

  // ---- epilogue: bias + optional relu + optional residual ----
  // C/D layout: VGPR r holds M = r + 8*half, N = lane%16
#pragma unroll
  for (int f = 0; f < 4; ++f) {
    int gn = col0 + f * 16 + ln;
    float bv = bias[gn];
#pragma unroll
    for (int r = 0; r < 8; ++r) {
      int gm = row0 + wv * 16 + half * 8 + r;
      if (gm < M) {
        float v = acc[f][r] + bv;
        if (relu) v = fmaxf(v, 0.f);
        size_t oidx = (size_t)gm * Ncols + gn;
        if (resid) v += resid[oidx];
        C[oidx] = v;
      }
    }
  }
}

// --------------------------------------------------------------------------
// agg[dst[e], :] += h[src[e], :]   (HID = 256, one thread = edge x 4 chans)
// h and agg both fit in the 192MB L2, so atomics resolve at L2.
// --------------------------------------------------------------------------
__global__ void scatter_add_kernel(const int* __restrict__ src,
                                   const int* __restrict__ dst,
                                   const float* __restrict__ h,
                                   float* __restrict__ agg, int E) {
  long long t = (long long)blockIdx.x * blockDim.x + threadIdx.x;
  long long total = (long long)E * 64;
  if (t >= total) return;
  int e = (int)(t >> 6);
  int c = (int)(t & 63) * 4;
  int s = src[e];
  int d = dst[e];
  float4 v = *(const float4*)(h + (size_t)s * 256 + c);
  float* out = agg + (size_t)d * 256 + c;
  atomicAdd(out + 0, v.x);
  atomicAdd(out + 1, v.y);
  atomicAdd(out + 2, v.z);
  atomicAdd(out + 3, v.w);
}

__global__ void zero_kernel(float* __restrict__ p, long long n4) {
  long long t = (long long)blockIdx.x * blockDim.x + threadIdx.x;
  if (t < n4) ((float4*)p)[t] = make_float4(0.f, 0.f, 0.f, 0.f);
}

__global__ void add_kernel(float* __restrict__ out, const float* __restrict__ a,
                           const float* __restrict__ b, long long n4) {
  long long t = (long long)blockIdx.x * blockDim.x + threadIdx.x;
  if (t < n4) {
    float4 x = ((const float4*)a)[t];
    float4 y = ((const float4*)b)[t];
    ((float4*)out)[t] = make_float4(x.x + y.x, x.y + y.y, x.z + y.z, x.w + y.w);
  }
}

// --------------------------------------------------------------------------
extern "C" void kernel_launch(void* const* d_in, const int* in_sizes, int n_in,
                              void* d_out, int out_size, void* d_ws, size_t ws_size,
                              hipStream_t stream) {
  (void)n_in; (void)out_size; (void)ws_size;

  const int IN_C = 128, HID = 256, OUT_C = 128, MLP_H = 512;

  const float* x    = (const float*)d_in[0];
  const int*   edge = (const int*)d_in[1];   // [2,E]: row0=src, row1=dst
  // d_in[2] = iterations (reference ignores it)
  const float* enc_w1 = (const float*)d_in[3];
  const float* enc_b1 = (const float*)d_in[4];
  const float* enc_w2 = (const float*)d_in[5];
  const float* enc_b2 = (const float*)d_in[6];
  const float* enc_w3 = (const float*)d_in[7];
  const float* enc_b3 = (const float*)d_in[8];
  const float* conv_w1 = (const float*)d_in[9];
  const float* conv_b1 = (const float*)d_in[10];
  const float* conv_w2 = (const float*)d_in[11];
  const float* conv_b2 = (const float*)d_in[12];
  const float* conv_w3 = (const float*)d_in[13];
  const float* conv_b3 = (const float*)d_in[14];
  const float* dec_w1 = (const float*)d_in[15];
  const float* dec_b1 = (const float*)d_in[16];
  const float* dec_w2 = (const float*)d_in[17];
  const float* dec_b2 = (const float*)d_in[18];
  const float* dec_w3 = (const float*)d_in[19];
  const float* dec_b3 = (const float*)d_in[20];

  const int M = in_sizes[0] / IN_C;   // node count
  const int E = in_sizes[1] / 2;      // edge count
  const int* e_src = edge;
  const int* e_dst = edge + E;

  // ---- workspace carve-out ----
  char* ws = (char*)d_ws;
  size_t off = 0;
  auto carve = [&](size_t bytes) -> void* {
    void* p = ws + off;
    off += (bytes + 255) & ~(size_t)255;
    return p;
  };
  float*  bufA = (float*)carve((size_t)M * MLP_H * sizeof(float));
  float*  bufB = (float*)carve((size_t)M * MLP_H * sizeof(float));
  float*  H    = (float*)carve((size_t)M * HID * sizeof(float));
  __bf16* wp_enc1  = (__bf16*)carve((size_t)IN_C * MLP_H * sizeof(__bf16));
  __bf16* wp_enc2  = (__bf16*)carve((size_t)MLP_H * MLP_H * sizeof(__bf16));
  __bf16* wp_enc3  = (__bf16*)carve((size_t)MLP_H * HID * sizeof(__bf16));
  __bf16* wp_conv1 = (__bf16*)carve((size_t)HID * MLP_H * sizeof(__bf16));
  __bf16* wp_conv2 = (__bf16*)carve((size_t)MLP_H * MLP_H * sizeof(__bf16));
  __bf16* wp_conv3 = (__bf16*)carve((size_t)MLP_H * HID * sizeof(__bf16));
  __bf16* wp_dec1  = (__bf16*)carve((size_t)HID * MLP_H * sizeof(__bf16));
  __bf16* wp_dec2  = (__bf16*)carve((size_t)MLP_H * MLP_H * sizeof(__bf16));
  __bf16* wp_dec3  = (__bf16*)carve((size_t)MLP_H * OUT_C * sizeof(__bf16));

  auto pack = [&](const float* w, __bf16* wp, int K, int N) {
    int total = K * N;
    pack_weight_bf16<<<(total + 255) / 256, 256, 0, stream>>>(w, wp, K, N);
  };
  auto gemm = [&](const float* A, const __bf16* Wp, const float* b,
                  const float* resid, float* Cout, int K, int Ncols, int relu) {
    dim3 grid((M + 63) / 64, Ncols / 64);
    gemm_wmma_bf16<<<grid, 128, 0, stream>>>(A, Wp, b, resid, Cout, M, K, Ncols, relu);
  };

  // ---- pack all 9 weight matrices to bf16 fragment order ----
  pack(enc_w1,  wp_enc1,  IN_C,  MLP_H);
  pack(enc_w2,  wp_enc2,  MLP_H, MLP_H);
  pack(enc_w3,  wp_enc3,  MLP_H, HID);
  pack(conv_w1, wp_conv1, HID,   MLP_H);
  pack(conv_w2, wp_conv2, MLP_H, MLP_H);
  pack(conv_w3, wp_conv3, MLP_H, HID);
  pack(dec_w1,  wp_dec1,  HID,   MLP_H);
  pack(dec_w2,  wp_dec2,  MLP_H, MLP_H);
  pack(dec_w3,  wp_dec3,  MLP_H, OUT_C);

  // ---- encoder: h = relu-MLP(x) ----
  gemm(x,    wp_enc1, enc_b1, nullptr, bufA, IN_C,  MLP_H, 1);
  gemm(bufA, wp_enc2, enc_b2, nullptr, bufB, MLP_H, MLP_H, 1);
  gemm(bufB, wp_enc3, enc_b3, nullptr, H,    MLP_H, HID,   1);

  // ---- GIN aggregation: agg (in bufA) = segment_sum(h[src] -> dst) ----
  long long n4 = (long long)M * HID / 4;
  zero_kernel<<<(unsigned)((n4 + 255) / 256), 256, 0, stream>>>(bufA, n4);
  long long sth = (long long)E * 64;
  scatter_add_kernel<<<(unsigned)((sth + 255) / 256), 256, 0, stream>>>(
      e_src, e_dst, H, bufA, E);
  // m = h + agg   (into bufA)
  add_kernel<<<(unsigned)((n4 + 255) / 256), 256, 0, stream>>>(bufA, bufA, H, n4);

  // ---- conv MLP with fused residual: h = h + relu-MLP(m) ----
  gemm(bufA, wp_conv1, conv_b1, nullptr, bufB, HID,   MLP_H, 1);
  gemm(bufB, wp_conv2, conv_b2, nullptr, bufA, MLP_H, MLP_H, 1);
  gemm(bufA, wp_conv3, conv_b3, H,       H,    MLP_H, HID,   1);  // resid fused

  // ---- decoder (no final activation) ----
  gemm(H,    wp_dec1, dec_b1, nullptr, bufA, HID,   MLP_H, 1);
  gemm(bufA, wp_dec2, dec_b2, nullptr, bufB, MLP_H, MLP_H, 1);
  gemm(bufB, wp_dec3, dec_b3, nullptr, (float*)d_out, MLP_H, OUT_C, 0);
}
